// GroupedQueryAttention_73280732004797
// MI455X (gfx1250) — matrix operand order repaired and checked by reading, not compiled, not run on hardware
//
#include <hip/hip_runtime.h>
#include <hip/hip_bf16.h>

// ---------------------------------------------------------------------------
// GQA forward: B=1, N=2048, D=2048, 16 Q heads / 4 KV heads, d=128, RoPE,
// causal softmax, out projection.  All GEMM-shaped work on bf16 WMMA
// (v_wmma_f32_16x16x32_bf16); GEMM B-panels staged in LDS via async loads
// (ASYNCcnt path); softmax stats in fp32.
// ---------------------------------------------------------------------------

#define N_TOK   2048
#define D_MODEL 2048
#define KV_D    512
#define HEAD_D  128

typedef __attribute__((ext_vector_type(16))) __bf16       v16bf;
typedef __attribute__((ext_vector_type(8)))  float        v8f;
typedef __attribute__((ext_vector_type(8)))  unsigned int v8u;
typedef int v4i __attribute__((vector_size(16)));   // matches builtin pointee

__device__ __forceinline__ unsigned short f2bf(float f) {
  unsigned int u = __builtin_bit_cast(unsigned int, f);
  u += 0x7FFFu + ((u >> 16) & 1u);   // round-to-nearest-even
  return (unsigned short)(u >> 16);
}

// A-matrix (16xK bf16) lane fragment: k = (e>=8 ? 16 : 0) + 8*half + (e&7).
// Caller passes (row base + kchunk + half*8); loads two 16B chunks.
__device__ __forceinline__ v16bf load_a_frag(const unsigned short* p) {
  uint4 a = *(const uint4*)p;
  uint4 b = *(const uint4*)(p + 16);
  v8u u = {a.x, a.y, a.z, a.w, b.x, b.y, b.z, b.w};
  return __builtin_bit_cast(v16bf, u);
}

// B-matrix (Kx16 bf16) lane fragment: k = (e>=8 ? 16 : 0) + 8*half + (e&7).
// Caller passes (column base + kchunk + half*8); loads two 16B chunks.
__device__ __forceinline__ v16bf load_b_frag(const unsigned short* p) {
  uint4 a = *(const uint4*)p;
  uint4 b = *(const uint4*)(p + 16);
  v8u u = {a.x, a.y, a.z, a.w, b.x, b.y, b.z, b.w};
  return __builtin_bit_cast(v16bf, u);
}

__device__ __forceinline__ v8f wmma_bf16(v16bf a, v16bf b, v8f c) {
  return __builtin_amdgcn_wmma_f32_16x16x32_bf16(false, a, false, b,
                                                 (short)0, c, false, false);
}

// ---- async global->LDS copy (16B per lane) + ASYNCcnt wait ----------------
#if defined(__has_builtin)
#if __has_builtin(__builtin_amdgcn_global_load_async_to_lds_b128)
#define ASYNC_CP_BUILTIN 1
#endif
#if __has_builtin(__builtin_amdgcn_s_wait_asynccnt)
#define WAIT_ASYNC_BUILTIN 1
#endif
#endif

__device__ __forceinline__ void async_copy16(unsigned short* lds,
                                             const unsigned short* g) {
#ifdef ASYNC_CP_BUILTIN
  __builtin_amdgcn_global_load_async_to_lds_b128(
      (__attribute__((address_space(1))) v4i*)g,
      (__attribute__((address_space(3))) v4i*)lds, 0, 0);
#else
  unsigned lofs = (unsigned)(size_t)lds;   // LDS aperture: low 32 bits = offset
  asm volatile("global_load_async_to_lds_b128 %0, %1, off"
               :: "v"(lofs), "v"(g) : "memory");
#endif
}

template <int Nleft>
__device__ __forceinline__ void wait_asynccnt() {
#ifdef WAIT_ASYNC_BUILTIN
  __builtin_amdgcn_s_wait_asynccnt(Nleft);
#else
  asm volatile("s_wait_asynccnt %0" :: "i"(Nleft) : "memory");
#endif
}

__device__ __forceinline__ float rowmax16(float v) {
  v = fmaxf(v, __shfl_xor(v, 1, 32));
  v = fmaxf(v, __shfl_xor(v, 2, 32));
  v = fmaxf(v, __shfl_xor(v, 4, 32));
  v = fmaxf(v, __shfl_xor(v, 8, 32));
  return v;
}
__device__ __forceinline__ float rowsum16(float v) {
  v += __shfl_xor(v, 1, 32);
  v += __shfl_xor(v, 2, 32);
  v += __shfl_xor(v, 4, 32);
  v += __shfl_xor(v, 8, 32);
  return v;
}

// ---------------------------------------------------------------------------
// Elementwise prep kernels
// ---------------------------------------------------------------------------
__global__ void convert_f32_bf16(const float* __restrict__ src,
                                 unsigned short* __restrict__ dst, int n) {
  int i = blockIdx.x * blockDim.x + threadIdx.x;
  if (i < n) dst[i] = f2bf(src[i]);
}

// W[K][N] fp32 -> Wt[N][K] bf16
__global__ void transpose_f32_bf16(const float* __restrict__ W,
                                   unsigned short* __restrict__ Wt,
                                   int K, int N) {
  int i = blockIdx.x * blockDim.x + threadIdx.x;
  if (i >= K * N) return;
  int k = i / N, nn = i - k * N;
  Wt[(size_t)nn * K + k] = f2bf(W[i]);
}

// RoPE on Q (channel layout g*512 + kv*128 + dd), fused 1/sqrt(128) scale.
__global__ void rope_q_kernel(const float* __restrict__ Qf,
                              unsigned short* __restrict__ Qb) {
  int idx = blockIdx.x * blockDim.x + threadIdx.x;  // N_TOK * 1024 pairs
  int t = idx >> 10;
  int ch = (idx & 1023) << 1;
  int dp = (ch & (HEAD_D - 1)) >> 1;                // pair index within head
  float ang = (float)t * __powf(10000.0f, -(float)(2 * dp) * (1.0f / 128.0f));
  float s, c;
  __sincosf(ang, &s, &c);
  size_t base = (size_t)t * D_MODEL + ch;
  float xe = Qf[base], xo = Qf[base + 1];
  const float sc = 0.08838834764831845f;            // 1/sqrt(128)
  Qb[base]     = f2bf((xe * c - xo * s) * sc);
  Qb[base + 1] = f2bf((xe * s + xo * c) * sc);
}

__global__ void rope_k_kernel(const float* __restrict__ Kf,
                              unsigned short* __restrict__ Kb) {
  int idx = blockIdx.x * blockDim.x + threadIdx.x;  // N_TOK * 256 pairs
  int t = idx >> 8;
  int ch = (idx & 255) << 1;
  int dp = (ch & (HEAD_D - 1)) >> 1;
  float ang = (float)t * __powf(10000.0f, -(float)(2 * dp) * (1.0f / 128.0f));
  float s, c;
  __sincosf(ang, &s, &c);
  size_t base = (size_t)t * KV_D + ch;
  float xe = Kf[base], xo = Kf[base + 1];
  Kb[base]     = f2bf(xe * c - xo * s);
  Kb[base + 1] = f2bf(xe * s + xo * c);
}

// V[N][512] fp32 -> Vt[512][N] bf16 (so P*V B-operand is contiguous in keys)
__global__ void transpose_v_kernel(const float* __restrict__ Vf,
                                   unsigned short* __restrict__ Vt) {
  int idx = blockIdx.x * blockDim.x + threadIdx.x;  // N_TOK * KV_D
  int t = idx >> 9;
  int ch = idx & 511;
  Vt[(size_t)ch * N_TOK + t] = f2bf(Vf[idx]);
}

// ---------------------------------------------------------------------------
// bf16 WMMA GEMM: C[MxN] fp32 = A[MxK] bf16 (row-major) @ Bt[N][K] bf16.
// Block tile 128x64: 8 waves x (16x64), all waves share one 64-column B panel
// staged in LDS by async global->LDS loads, double buffered over 32-k steps.
// Panel rows padded to 40 elems (80B stride -> conflict-free 16-lane reads).
// M%128==0, N%64==0, K%32==0.
// ---------------------------------------------------------------------------
#define BPAD 40

__global__ __launch_bounds__(256) void gemm_bf16_wmma(
    const unsigned short* __restrict__ A, const unsigned short* __restrict__ Bt,
    float* __restrict__ C, int M, int N, int K) {
  __shared__ __align__(16) unsigned short Bpanel[2][64 * BPAD];
  const int tid  = threadIdx.x;
  const int lane = tid & 31;
  const int n    = lane & 15;
  const int half = lane >> 4;
  const int wv   = tid >> 5;                 // wave 0..7 -> 16-row slice
  const int ntiles = N >> 6;
  const int bm = blockIdx.x / ntiles;        // 128-row supertile
  const int col0 = (blockIdx.x - bm * ntiles) << 6;

  // per-thread async slice of the 64x32 bf16 panel: 16B each
  const int pc = tid >> 2;                   // column 0..63
  const int sg = (tid & 3) * 8;              // k-segment offset (elements)
  const unsigned short* gsrc = Bt + (size_t)(col0 + pc) * K + sg;
  unsigned short* ldst[2] = {&Bpanel[0][pc * BPAD + sg], &Bpanel[1][pc * BPAD + sg]};

  const unsigned short* arow = A + (size_t)(bm * 128 + wv * 16 + n) * K;
  v8f acc[4] = {};

  async_copy16(ldst[0], gsrc);               // prefetch k=0 panel
  for (int k = 0; k < K; k += 32) {
    const int buf = (k >> 5) & 1;
    if (k + 32 < K) {
      async_copy16(ldst[buf ^ 1], gsrc + k + 32);
      wait_asynccnt<1>();                    // current panel landed in LDS
    } else {
      wait_asynccnt<0>();
    }
    __syncthreads();                         // panel visible to all waves

    v16bf a = load_a_frag(arow + k + half * 8);
    const unsigned short* panel = &Bpanel[buf][0];
#pragma unroll
    for (int j = 0; j < 4; ++j) {
      v16bf b = load_b_frag(panel + (j * 16 + n) * BPAD + half * 8);
      acc[j] = wmma_bf16(a, b, acc[j]);
    }
    __syncthreads();                         // done reading before overwrite
  }

#pragma unroll
  for (int j = 0; j < 4; ++j)
#pragma unroll
    for (int r = 0; r < 8; ++r)
      C[(size_t)(bm * 128 + wv * 16 + r + 8 * half) * N + col0 + j * 16 + n] =
          acc[j][r];
}

// ---------------------------------------------------------------------------
// Flash attention: one wave per (head, 16-row Q block). 32 keys per step:
// S = Q*K^T (8 WMMAs), online softmax (cross-lane row reductions), P restriped
// through per-wave LDS tile into A-layout, O += P*V (8 WMMAs).
// ---------------------------------------------------------------------------
__global__ __launch_bounds__(256) void attn_kernel(
    const unsigned short* __restrict__ Qb, const unsigned short* __restrict__ Kb,
    const unsigned short* __restrict__ Vt, unsigned short* __restrict__ Cb) {
  __shared__ __align__(16) unsigned short Plds[8][512];  // 16x32 bf16 per wave
  const int lane = threadIdx.x & 31;
  const int n    = lane & 15;
  const int half = lane >> 4;
  const int w    = threadIdx.x >> 5;
  const int wave = blockIdx.x * 8 + w;
  const int head = wave >> 7;           // 0..15 = kv*4 + g
  const int q0   = (wave & 127) << 4;
  const int kv = head >> 2, g = head & 3;
  const int qc = g * 512 + kv * HEAD_D; // Q channel base (gs-major layout)
  const int kc = kv * HEAD_D;           // K/V channel base
  unsigned short* P = &Plds[w][0];

  // Resident Q fragments (16x128 = 4 x 16x32 A-fragments), pre-scaled by 1/sqrt(d)
  const unsigned short* qrow = Qb + (size_t)(q0 + n) * D_MODEL + qc;
  v16bf qf[4];
#pragma unroll
  for (int c = 0; c < 4; ++c) qf[c] = load_a_frag(qrow + c * 32 + half * 8);

  v8f O[8] = {};
  float m[8], l[8];
#pragma unroll
  for (int r = 0; r < 8; ++r) { m[r] = -3.0e38f; l[r] = 0.0f; }

  for (int kb = 0; kb <= q0 + 15; kb += 32) {   // last kb<=2016 -> keys<=2047
    v8f s0 = {}, s1 = {};
    const unsigned short* k0row = Kb + (size_t)(kb + n) * KV_D + kc;
    const unsigned short* k1row = Kb + (size_t)(kb + 16 + n) * KV_D + kc;
#pragma unroll
    for (int c = 0; c < 4; ++c) {
      s0 = wmma_bf16(qf[c], load_b_frag(k0row + c * 32 + half * 8), s0);
      s1 = wmma_bf16(qf[c], load_b_frag(k1row + c * 32 + half * 8), s1);
    }

    const int key0 = kb + n, key1 = kb + 16 + n;
    float alpha[8];
#pragma unroll
    for (int r = 0; r < 8; ++r) {
      const int qrg = q0 + r + 8 * half;        // global query row of this slot
      float v0 = (key0 <= qrg) ? s0[r] : -3.0e38f;
      float v1 = (key1 <= qrg) ? s1[r] : -3.0e38f;
      float mr = rowmax16(fmaxf(v0, v1));
      float mn = fmaxf(m[r], mr);
      alpha[r] = __expf(m[r] - mn);
      m[r] = mn;
      float p0 = __expf(v0 - mn);
      float p1 = __expf(v1 - mn);
      l[r] = l[r] * alpha[r] + rowsum16(p0 + p1);
      P[(r + 8 * half) * 32 + n]      = f2bf(p0);   // C-layout -> [row][key] LDS
      P[(r + 8 * half) * 32 + 16 + n] = f2bf(p1);
    }
#pragma unroll
    for (int c = 0; c < 8; ++c)
#pragma unroll
      for (int r = 0; r < 8; ++r) O[c][r] *= alpha[r];

    asm volatile("s_wait_dscnt 0x0" ::: "memory");  // LDS store->load ordering
    v16bf pf = load_a_frag(P + n * 32 + half * 8);  // P as 16x32 A-fragment

#pragma unroll
    for (int c = 0; c < 8; ++c) {
      const unsigned short* vrow =
          Vt + (size_t)(kc + c * 16 + n) * N_TOK + kb + half * 8;
      O[c] = wmma_bf16(pf, load_b_frag(vrow), O[c]);
    }
  }

  // Normalize and store ctx in flattened head order (kv-major: head*128 + dd)
#pragma unroll
  for (int r = 0; r < 8; ++r) {
    const float inv = 1.0f / l[r];
    unsigned short* crow =
        Cb + (size_t)(q0 + r + 8 * half) * D_MODEL + head * HEAD_D + n;
#pragma unroll
    for (int c = 0; c < 8; ++c) crow[c * 16] = f2bf(O[c][r] * inv);
  }
}

// ---------------------------------------------------------------------------
// Host launch
// ---------------------------------------------------------------------------
extern "C" void kernel_launch(void* const* d_in, const int* in_sizes, int n_in,
                              void* d_out, int out_size, void* d_ws, size_t ws_size,
                              hipStream_t stream) {
  (void)in_sizes; (void)n_in; (void)out_size; (void)ws_size;
  const float* x  = (const float*)d_in[0];
  const float* Wq = (const float*)d_in[1];
  const float* Wk = (const float*)d_in[2];
  const float* Wv = (const float*)d_in[3];
  const float* Wo = (const float*)d_in[4];
  float* out = (float*)d_out;

  char* ws = (char*)d_ws;
  const size_t MB = 1ull << 20;
  unsigned short* xb  = (unsigned short*)(ws + 0 * MB);   // 8 MB
  unsigned short* Wqt = (unsigned short*)(ws + 8 * MB);   // 8 MB
  unsigned short* Wkt = (unsigned short*)(ws + 16 * MB);  // 2 MB
  unsigned short* Wvt = (unsigned short*)(ws + 18 * MB);  // 2 MB
  unsigned short* Wot = (unsigned short*)(ws + 20 * MB);  // 8 MB
  unsigned short* Qbf = (unsigned short*)(ws + 28 * MB);  // 8 MB
  unsigned short* Kbf = (unsigned short*)(ws + 36 * MB);  // 2 MB
  unsigned short* Vtb = (unsigned short*)(ws + 38 * MB);  // 2 MB
  unsigned short* Cb  = (unsigned short*)(ws + 40 * MB);  // 8 MB
  float* Qf = (float*)(ws + 48 * MB);                     // 16 MB
  float* Kf = (float*)(ws + 64 * MB);                     // 4 MB
  float* Vf = (float*)(ws + 68 * MB);                     // 4 MB  (total 72 MB)

  const int thr = 256;
  convert_f32_bf16<<<(N_TOK * D_MODEL) / thr, thr, 0, stream>>>(x, xb, N_TOK * D_MODEL);
  transpose_f32_bf16<<<(D_MODEL * D_MODEL) / thr, thr, 0, stream>>>(Wq, Wqt, D_MODEL, D_MODEL);
  transpose_f32_bf16<<<(D_MODEL * KV_D) / thr, thr, 0, stream>>>(Wk, Wkt, D_MODEL, KV_D);
  transpose_f32_bf16<<<(D_MODEL * KV_D) / thr, thr, 0, stream>>>(Wv, Wvt, D_MODEL, KV_D);
  transpose_f32_bf16<<<(D_MODEL * D_MODEL) / thr, thr, 0, stream>>>(Wo, Wot, D_MODEL, D_MODEL);

  // GEMM grid: (M/128) * (N/64) blocks of 256 threads
  gemm_bf16_wmma<<<16 * 32, 256, 0, stream>>>(xb, Wqt, Qf, N_TOK, D_MODEL, D_MODEL);
  gemm_bf16_wmma<<<16 * 8, 256, 0, stream>>>(xb, Wkt, Kf, N_TOK, KV_D, D_MODEL);
  gemm_bf16_wmma<<<16 * 8, 256, 0, stream>>>(xb, Wvt, Vf, N_TOK, KV_D, D_MODEL);

  rope_q_kernel<<<(N_TOK * 1024) / thr, thr, 0, stream>>>(Qf, Qbf);
  rope_k_kernel<<<(N_TOK * 256) / thr, thr, 0, stream>>>(Kf, Kbf);
  transpose_v_kernel<<<(N_TOK * KV_D) / thr, thr, 0, stream>>>(Vf, Vtb);

  // 16 heads x 128 q-blocks = 2048 waves, 8 waves/block
  attn_kernel<<<256, 256, 0, stream>>>(Qbf, Kbf, Vtb, Cb);

  // Output projection -> fp32 d_out
  gemm_bf16_wmma<<<16 * 32, 256, 0, stream>>>(Cb, Wot, out, N_TOK, D_MODEL, D_MODEL);
}
